// GatingNetwork_4243427688928
// MI455X (gfx1250) — matrix operand order repaired and checked by reading, not compiled, not run on hardware
//
#include <hip/hip_runtime.h>
#include <hip/hip_bf16.h>
#include <math.h>

// ---------------------------------------------------------------------------
// GatingNetwork for MI455X (gfx1250, wave32, WMMA).
//   x = concat(hist[50], num[32], cat_emb[8x16], region_emb[32]) -> 242 (pad 256)
//   h1 = gelu(x @ W1 + b1)   [242x64]  -> WMMA f16, K-steps = 8
//   h2 = gelu(h1 @ W2 + b2)  [64x64]   -> WMMA f16, K-steps = 2
//   lg = h2 @ W3 + b3        [64x64]   -> WMMA f16, K-steps = 2
//   alpha = softmax(top8-mask(lg))
// One 256-thread block (8 wave32) handles 32 rows; each wave owns a 16x16 tile.
// GELU uses a branchless erf approximation (A&S 7.1.26, |err| < 1.5e-7) built
// on the native v_exp_f32 TRANS op instead of libm's branchy erff.
// ---------------------------------------------------------------------------

typedef _Float16 v16h __attribute__((ext_vector_type(16)));
typedef _Float16 h8   __attribute__((ext_vector_type(8)));
typedef float    v8f  __attribute__((ext_vector_type(8)));

#define BROWS   32
#define KPAD    256
#define HIDDEN  64
#define NEXP    64
#define TOPK    8

// A-fragment (16x32 f16, M x K) from row-major LDS [rows][stride].
// Lane half h: holds K = ko + h*16 .. +15 (contiguous), M = lane&15.
__device__ __forceinline__ v16h load_a_frag(const _Float16* base, int mbase,
                                            int stride, int ko, int lane) {
    const int h = (lane >> 4) & 1;
    const int m = lane & 15;
    const _Float16* row = base + (mbase + m) * stride + ko;
    return *(const v16h*)(row + h * 16);
}

// B-fragment (32x16 f16, K x N) from column-major LDS weights [n][stridek].
// Lane half h: holds K = ko + h*16 .. +15 (contiguous), N = lane&15.
__device__ __forceinline__ v16h load_b_frag(const _Float16* wb, int nbase,
                                            int stridek, int ko, int lane) {
    const int h = (lane >> 4) & 1;
    const int n = lane & 15;
    const _Float16* p = wb + (nbase + n) * stridek + ko + h * 16;
    return *(const v16h*)p;
}

// Branchless erf (Abramowitz & Stegun 7.1.26): |abs err| < 1.5e-7.
// One divide (v_rcp + refine), one native v_exp_f32, 6 FMAs, no EXEC branching.
__device__ __forceinline__ float erf_fast(float x) {
    const float ax = fabsf(x);
    const float t  = 1.0f / fmaf(0.3275911f, ax, 1.0f);
    float p = 1.061405429f;
    p = fmaf(p, t, -1.453152027f);
    p = fmaf(p, t,  1.421413741f);
    p = fmaf(p, t, -0.284496736f);
    p = fmaf(p, t,  0.254829592f);
    const float e = __expf(-ax * ax);          // v_exp_f32
    const float r = fmaf(-p * t, e, 1.0f);
    return copysignf(r, x);
}

__device__ __forceinline__ float gelu_exact(float x) {
    return 0.5f * x * (1.0f + erf_fast(x * 0.70710678118654752f));
}

__global__ __launch_bounds__(256) void gating_moe_kernel(
    const float* __restrict__ hist,      // [B,50]
    const float* __restrict__ num,       // [B,32]
    const int*   __restrict__ cat,       // [B,8]
    const int*   __restrict__ region,    // [B]
    const float* __restrict__ cat_tab,   // [8,32768,16]
    const float* __restrict__ reg_tab,   // [4096,32]
    const float* __restrict__ W1,        // [242,64]
    const float* __restrict__ b1,        // [64]
    const float* __restrict__ W2,        // [64,64]
    const float* __restrict__ b2,        // [64]
    const float* __restrict__ W3,        // [64,64]
    const float* __restrict__ b3,        // [64]
    float* __restrict__ out,             // [B,64]
    int Btotal)
{
    // LDS layout (56 KB): logits alias the X buffer (X dead after layer 1).
    __shared__ __align__(64) unsigned char smem[57344];
    _Float16* xbuf  = (_Float16*)(smem);           // 32 x 256 f16 (16 KB)
    _Float16* wbuf  = (_Float16*)(smem + 16384);   // W1: 64 x 256 f16 (32 KB);
                                                   // later W2 @ +0, W3 @ +4096 halves
    _Float16* hbuf  = (_Float16*)(smem + 49152);   // 32 x 64 f16 (4 KB)
    _Float16* h2buf = (_Float16*)(smem + 53248);   // 32 x 64 f16 (4 KB)
    float*    lbuf  = (float*)(smem);              // 32 x 64 f32 (8 KB, aliases xbuf)

    const int tid  = threadIdx.x;
    const int lane = tid & 31;
    const int wave = tid >> 5;
    const long row0 = (long)blockIdx.x * BROWS;

    // ---------------- Phase 0: stage X tile (with gathers) + W1 ----------------
    for (int idx = tid; idx < BROWS * KPAD; idx += 256) {
        const int r = idx >> 8;
        const int c = idx & 255;   // c == tid: branch pattern is loop-invariant
        long grow = row0 + r;
        if (grow >= Btotal) grow = Btotal - 1;
        float v;
        if (c < 50) {
            v = hist[grow * 50 + c];
        } else if (c < 82) {
            v = num[grow * 32 + (c - 50)];
        } else if (c < 210) {
            const int cc = c - 82;
            const int t  = cc >> 4;
            const int d  = cc & 15;
            int ix = cat[grow * 8 + t];
            ix = ix < 0 ? 0 : (ix > 32767 ? 32767 : ix);
            v = cat_tab[((long)t * 32768 + ix) * 16 + d];
        } else if (c < 242) {
            int rid = region[grow];
            rid = rid < 0 ? 0 : (rid > 4095 ? 4095 : rid);
            v = reg_tab[(long)rid * 32 + (c - 210)];
        } else {
            v = 0.0f;   // K padding 242 -> 256
        }
        xbuf[idx] = (_Float16)v;
    }
    // W1 column-major [n][256], zero-padded rows 242..255
    for (int idx = tid; idx < HIDDEN * KPAD; idx += 256) {
        const int n = idx >> 8;
        const int k = idx & 255;
        const float v = (k < 242) ? W1[(long)k * HIDDEN + n] : 0.0f;
        wbuf[idx] = (_Float16)v;
    }
    __syncthreads();

    // Wave -> 16x16 output tile
    const int mt = (wave >> 2) * 16;  // 0 | 16
    const int nt = (wave & 3) * 16;   // 0 | 16 | 32 | 48
    const int cn  = nt + (lane & 15);        // this lane's output column
    const int mhi = ((lane >> 4) & 1) * 8;   // C-matrix M offset per lane half

    // ---------------- Layer 1: [32x256] x [256x64] ----------------
    v8f acc = {};
#pragma unroll
    for (int ko = 0; ko < KPAD; ko += 32) {
        v16h a = load_a_frag(xbuf, mt, KPAD, ko, lane);
        v16h b = load_b_frag(wbuf, nt, KPAD, ko, lane);
        acc = __builtin_amdgcn_wmma_f32_16x16x32_f16(
            false, a, false, b, (short)0, acc, false, false);
    }
    {
        const float bias = b1[cn];
#pragma unroll
        for (int v = 0; v < 8; ++v) {
            const float g = gelu_exact(acc[v] + bias);
            hbuf[(mt + v + mhi) * HIDDEN + cn] = (_Float16)g;
        }
    }
    __syncthreads();

    // ---------------- Restage W2, W3 (column-major [n][64]) ----------------
    for (int idx = tid; idx < HIDDEN * HIDDEN; idx += 256) {
        const int n = idx >> 6;
        const int k = idx & 63;
        wbuf[idx]        = (_Float16)W2[(long)k * HIDDEN + n];
        wbuf[4096 + idx] = (_Float16)W3[(long)k * HIDDEN + n];
    }
    __syncthreads();

    // ---------------- Layer 2: [32x64] x [64x64] ----------------
    acc = (v8f){};
#pragma unroll
    for (int ko = 0; ko < HIDDEN; ko += 32) {
        v16h a = load_a_frag(hbuf, mt, HIDDEN, ko, lane);
        v16h b = load_b_frag(wbuf, nt, HIDDEN, ko, lane);
        acc = __builtin_amdgcn_wmma_f32_16x16x32_f16(
            false, a, false, b, (short)0, acc, false, false);
    }
    {
        const float bias = b2[cn];
#pragma unroll
        for (int v = 0; v < 8; ++v) {
            const float g = gelu_exact(acc[v] + bias);
            h2buf[(mt + v + mhi) * HIDDEN + cn] = (_Float16)g;
        }
    }
    __syncthreads();

    // ---------------- Layer 3: logits = h2 @ W3 + b3 (f32 in LDS) ----------------
    acc = (v8f){};
#pragma unroll
    for (int ko = 0; ko < HIDDEN; ko += 32) {
        v16h a = load_a_frag(h2buf, mt, HIDDEN, ko, lane);
        v16h b = load_b_frag(wbuf + 4096, nt, HIDDEN, ko, lane);
        acc = __builtin_amdgcn_wmma_f32_16x16x32_f16(
            false, a, false, b, (short)0, acc, false, false);
    }
    {
        const float bias = b3[cn];
#pragma unroll
        for (int v = 0; v < 8; ++v) {
            lbuf[(mt + v + mhi) * NEXP + cn] = acc[v] + bias;
        }
    }
    __syncthreads();

    // ---------------- Top-8 mask + softmax (4 rows per wave, 2 logits/lane) ------
#pragma unroll
    for (int rr = 0; rr < 4; ++rr) {
        const int r = wave * 4 + rr;
        const float* lrow = lbuf + r * NEXP;
        const int i0 = lane, i1 = lane + 32;
        const float v0 = lrow[i0];
        const float v1 = lrow[i1];
        // exact top-k rank: count strictly-greater, tie-break by smaller index
        int rank0 = 0, rank1 = 0;
#pragma unroll 4
        for (int j = 0; j < NEXP; ++j) {
            const float vj = lrow[j];   // LDS broadcast read
            rank0 += (vj > v0) || (vj == v0 && j < i0);
            rank1 += (vj > v1) || (vj == v1 && j < i1);
        }
        const bool keep0 = rank0 < TOPK;
        const bool keep1 = rank1 < TOPK;
        // wave32 max reduce (global max is always in the top-8)
        float m = fmaxf(v0, v1);
#pragma unroll
        for (int off = 16; off > 0; off >>= 1) m = fmaxf(m, __shfl_xor(m, off, 32));
        const float e0 = keep0 ? __expf(v0 - m) : 0.0f;   // v_exp_f32
        const float e1 = keep1 ? __expf(v1 - m) : 0.0f;
        float s = e0 + e1;
#pragma unroll
        for (int off = 16; off > 0; off >>= 1) s += __shfl_xor(s, off, 32);
        const float inv = 1.0f / s;
        const long grow = row0 + r;
        if (grow < Btotal) {
            out[grow * NEXP + i0] = e0 * inv;
            out[grow * NEXP + i1] = e1 * inv;
        }
    }
}

extern "C" void kernel_launch(void* const* d_in, const int* in_sizes, int n_in,
                              void* d_out, int out_size, void* d_ws, size_t ws_size,
                              hipStream_t stream) {
    const float* hist    = (const float*)d_in[0];
    const float* num     = (const float*)d_in[1];
    const int*   cat     = (const int*)  d_in[2];
    const int*   region  = (const int*)  d_in[3];
    const float* cat_tab = (const float*)d_in[4];
    const float* reg_tab = (const float*)d_in[5];
    const float* W1      = (const float*)d_in[6];
    const float* b1      = (const float*)d_in[7];
    const float* W2      = (const float*)d_in[8];
    const float* b2      = (const float*)d_in[9];
    const float* W3      = (const float*)d_in[10];
    const float* b3      = (const float*)d_in[11];
    float* out = (float*)d_out;

    const int B = in_sizes[0] / 50;               // hist_y is [B, 50]
    const int blocks = (B + BROWS - 1) / BROWS;   // 16384 for B = 524288

    gating_moe_kernel<<<blocks, 256, 0, stream>>>(
        hist, num, cat, region, cat_tab, reg_tab,
        W1, b1, W2, b2, W3, b3, out, B);
}
